// TransformerBlock_68006512164948
// MI455X (gfx1250) — matrix operand order repaired and checked by reading, not compiled, not run on hardware
//
#include <hip/hip_runtime.h>
#include <stdint.h>

// ---------- types ----------
typedef __attribute__((ext_vector_type(16))) __bf16    v16bf;
typedef __attribute__((ext_vector_type(8)))  float     v8f;
typedef __attribute__((ext_vector_type(4)))  uint32_t  v4u;

union Frag16 { v4u q[2]; v16bf v; };   // 32 bytes: one 16-bit WMMA operand (16 bf16 / lane)

__device__ __forceinline__ uint16_t f32_to_bf16(float f) {
  uint32_t u = __float_as_uint(f);
  uint32_t r = u + 0x7FFFu + ((u >> 16) & 1u);   // round-to-nearest-even
  return (uint16_t)(r >> 16);
}

__device__ __forceinline__ v8f vzero8() {
  v8f z;
  #pragma unroll
  for (int i = 0; i < 8; ++i) z[i] = 0.f;
  return z;
}

__device__ __forceinline__ v8f wmma_bf16(v16bf a, v16bf b, v8f c) {
  // D = A(16x32 bf16) * B(32x16 bf16) + C(16x16 f32)
  return __builtin_amdgcn_wmma_f32_16x16x32_bf16(false, a, false, b, (short)0, c, false, false);
}

// Per-lane async DMA: 16 bytes global -> LDS, tracked by ASYNCcnt (no VGPR staging).
// LDS byte address = low 32 bits of the generic pointer to a __shared__ object.
__device__ __forceinline__ void async_ld_lds_b128(const void* gptr, uint32_t lds_addr) {
  asm volatile("global_load_async_to_lds_b128 %0, %1, off"
               :: "v"(lds_addr), "v"(gptr)
               : "memory");
}

// ASYNCcnt decrements in order: waiting <=8 guarantees the *previous* 8-transfer
// tile has landed while the most recent 8 (next tile) may still be in flight.
__device__ __forceinline__ void wait_asynccnt0() {
  asm volatile("s_wait_asynccnt 0" ::: "memory");
}
__device__ __forceinline__ void wait_asynccnt8() {
  asm volatile("s_wait_asynccnt 8" ::: "memory");
}

// ---------- elementwise ----------
__global__ __launch_bounds__(256)
void k_f32_to_bf16(const float* __restrict__ src, uint16_t* __restrict__ dst, size_t n) {
  size_t i = (size_t)blockIdx.x * 256 + threadIdx.x;
  if (i < n) dst[i] = f32_to_bf16(src[i]);
}

__global__ __launch_bounds__(256)
void k_rmsnorm_bf16(const float* __restrict__ x, const float* __restrict__ g,
                    uint16_t* __restrict__ out) {
  const int row = blockIdx.x;
  const float* xr = x + (size_t)row * 1024;
  float ss = 0.f;
  for (int i = threadIdx.x; i < 1024; i += 256) { float v = xr[i]; ss += v * v; }
  #pragma unroll
  for (int off = 1; off < 32; off <<= 1) ss += __shfl_xor(ss, off, 32);
  __shared__ float red[8];
  if ((threadIdx.x & 31) == 0) red[threadIdx.x >> 5] = ss;
  __syncthreads();
  float tot = 0.f;
  #pragma unroll
  for (int w = 0; w < 8; ++w) tot += red[w];
  const float rs = rsqrtf(tot * (1.f / 1024.f) + 1e-6f);
  for (int i = threadIdx.x; i < 1024; i += 256)
    out[(size_t)row * 1024 + i] = f32_to_bf16(xr[i] * g[i] * rs);
}

// qkv f32 [B*T, 3*1024] -> rope(q,k) -> q/k/v bf16 [B,H,T,64]
__global__ __launch_bounds__(256)
void k_rope_split(const float* __restrict__ qkv, const float* __restrict__ cs,
                  const float* __restrict__ sn, uint16_t* __restrict__ q,
                  uint16_t* __restrict__ k, uint16_t* __restrict__ v, int T, size_t n) {
  size_t i = (size_t)blockIdx.x * 256 + threadIdx.x;
  if (i >= n) return;
  const int hd = (int)(i & 63);
  const int h  = (int)((i >> 6) & 15);
  const int s  = (int)((i / 1024) % 3);
  const size_t row = i / 3072;
  const int t = (int)(row % (size_t)T);
  const int b = (int)(row / (size_t)T);
  const float val = qkv[i];
  float o = val;
  if (s < 2) {
    const float pv = qkv[(hd < 32) ? (i + 32) : (i - 32)];
    const float rot = (hd < 32) ? -pv : pv;
    o = val * cs[t * 64 + hd] + rot * sn[t * 64 + hd];
  }
  const size_t dst = (((size_t)(b * 16 + h) * T) + t) * 64 + hd;
  const uint16_t hb = f32_to_bf16(o);
  if (s == 0) q[dst] = hb; else if (s == 1) k[dst] = hb; else v[dst] = hb;
}

// f32 [B*T, 1024] -> bf16 [B,H,T,64]
__global__ __launch_bounds__(256)
void k_split_heads(const float* __restrict__ src, uint16_t* __restrict__ dst, int T, size_t n) {
  size_t i = (size_t)blockIdx.x * 256 + threadIdx.x;
  if (i >= n) return;
  const int hd = (int)(i & 63);
  const int h  = (int)((i >> 6) & 15);
  const size_t row = i / 1024;
  const int t = (int)(row % (size_t)T);
  const int b = (int)(row / (size_t)T);
  dst[(((size_t)(b * 16 + h) * T) + t) * 64 + hd] = f32_to_bf16(src[i]);
}

// f32 [B*S, 2*1024] -> k/v bf16 [B,H,S,64]
__global__ __launch_bounds__(256)
void k_split_kv(const float* __restrict__ src, uint16_t* __restrict__ kd,
                uint16_t* __restrict__ vd, int S, size_t n) {
  size_t i = (size_t)blockIdx.x * 256 + threadIdx.x;
  if (i >= n) return;
  const int hd = (int)(i & 63);
  const int h  = (int)((i >> 6) & 15);
  const int s2 = (int)((i / 1024) & 1);
  const size_t row = i / 2048;
  const int s = (int)(row % (size_t)S);
  const int b = (int)(row / (size_t)S);
  const size_t dst = (((size_t)(b * 16 + h) * S) + s) * 64 + hd;
  const uint16_t hb = f32_to_bf16(src[i]);
  if (s2 == 0) kd[dst] = hb; else vd[dst] = hb;
}

// ---------- WMMA GEMM: C[M,N] = A[M,K] * W[N,K]^T (+epilogue) ----------
// Double-buffered LDS: tile k+1 DMA'd via GLOBAL_LOAD_ASYNC_TO_LDS_B128 while
// tile k feeds the WMMA stream; partial s_wait_asynccnt keeps the overlap.
// EPI 0: Cf = acc ; EPI 1: Cf = acc + aux ; EPI 2: Cb = bf16(silu(aux) * acc)
template<int EPI>
__global__ __launch_bounds__(256)
void k_gemm_bf16(const uint16_t* __restrict__ A, const uint16_t* __restrict__ W,
                 float* __restrict__ Cf, uint16_t* __restrict__ Cb,
                 const float* __restrict__ aux, int M, int N, int K) {
  constexpr int BM = 128, BN = 128, BK = 64;
  __shared__ uint16_t As[2][BM * BK];
  __shared__ uint16_t Bs[2][BN * BK];
  const int tid  = threadIdx.x;
  const int lane = tid & 31;
  const int w    = tid >> 5;            // 8 waves
  const int wm   = (w >> 1) * 32;       // wave tile: 32 x 64
  const int wn   = (w & 1) * 64;
  const int m0   = blockIdx.y * BM;
  const int n0   = blockIdx.x * BN;
  const int lrow = lane & 15;
  const int lhi  = lane >> 4;

  v8f acc[2][4];
  #pragma unroll
  for (int mt = 0; mt < 2; ++mt)
    #pragma unroll
    for (int nt = 0; nt < 4; ++nt) acc[mt][nt] = vzero8();

  // stage a 128x64 bf16 tile pair (A and W) into LDS buffer `buf` via async DMA
  auto stage = [&](int buf, int k0) {
    #pragma unroll
    for (int i = 0; i < 4; ++i) {
      const int vecid = tid + i * 256;          // 0..1023
      const int row = vecid >> 3;               // 8 x 16B per row
      const int c8  = (vecid & 7) * 8;
      const uint16_t* ga = &A[(size_t)(m0 + row) * K + k0 + c8];
      const uint16_t* gw = &W[(size_t)(n0 + row) * K + k0 + c8];
      async_ld_lds_b128(ga, (uint32_t)(uintptr_t)&As[buf][row * BK + c8]);
      async_ld_lds_b128(gw, (uint32_t)(uintptr_t)&Bs[buf][row * BK + c8]);
      if (k0 + BK < K) {                        // warm L2 one tile ahead of the DMA
        __builtin_prefetch(ga + BK, 0, 1);
        __builtin_prefetch(gw + BK, 0, 1);
      }
    }
  };

  stage(0, 0);
  int buf = 0;
  for (int k0 = 0; k0 < K; k0 += BK) {
    const bool have_next = (k0 + BK) < K;
    if (have_next) {
      stage(buf ^ 1, k0 + BK);   // next tile DMA in flight during this tile's WMMAs
      wait_asynccnt8();          // only current tile's 8 transfers must have landed
    } else {
      wait_asynccnt0();
    }
    __syncthreads();
    #pragma unroll
    for (int ks = 0; ks < BK; ks += 32) {
      Frag16 fa[2], fb[4];
      #pragma unroll
      for (int mt = 0; mt < 2; ++mt) {
        const uint16_t* p = &As[buf][(wm + mt * 16 + lrow) * BK + ks + lhi * 8];
        fa[mt].q[0] = *(const v4u*)p;
        fa[mt].q[1] = *(const v4u*)(p + 16);
      }
      #pragma unroll
      for (int nt = 0; nt < 4; ++nt) {
        const uint16_t* p = &Bs[buf][(wn + nt * 16 + lrow) * BK + ks + lhi * 8];
        fb[nt].q[0] = *(const v4u*)p;
        fb[nt].q[1] = *(const v4u*)(p + 16);
      }
      #pragma unroll
      for (int mt = 0; mt < 2; ++mt)
        #pragma unroll
        for (int nt = 0; nt < 4; ++nt)
          acc[mt][nt] = wmma_bf16(fa[mt].v, fb[nt].v, acc[mt][nt]);
    }
    __syncthreads();
    buf ^= 1;
  }

  #pragma unroll
  for (int mt = 0; mt < 2; ++mt)
    #pragma unroll
    for (int nt = 0; nt < 4; ++nt)
      #pragma unroll
      for (int r = 0; r < 8; ++r) {
        const int gm = m0 + wm + mt * 16 + lhi * 8 + r;
        const int gn = n0 + wn + nt * 16 + lrow;
        const size_t idx = (size_t)gm * N + gn;
        const float v = acc[mt][nt][r];
        if (EPI == 0) {
          Cf[idx] = v;
        } else if (EPI == 1) {
          Cf[idx] = v + aux[idx];
        } else {
          const float a0 = aux[idx];
          const float si = a0 / (1.f + __expf(-a0));
          Cb[idx] = f32_to_bf16(si * v);
        }
      }
}

// ---------- WMMA flash attention: 1 wave / (b,h,16-query tile) ----------
template<bool CAUSAL>
__global__ __launch_bounds__(32)
void k_attn_wmma(const uint16_t* __restrict__ Qh, const uint16_t* __restrict__ Kh,
                 const uint16_t* __restrict__ Vh, const int* __restrict__ cmask,
                 uint16_t* __restrict__ Out, int T, int S, float rscale) {
  const int tiles = T >> 4;
  const int bh = blockIdx.x / tiles;
  const int t0 = (blockIdx.x - bh * tiles) << 4;
  const int b  = bh >> 4;
  const int h  = bh & 15;
  const int lane = threadIdx.x;
  const int lrow = lane & 15;
  const int lhi  = lane >> 4;

  const uint16_t* Qb = Qh + (size_t)bh * T * 64;
  const uint16_t* Kb = Kh + (size_t)bh * S * 64;
  const uint16_t* Vb = Vh + (size_t)bh * S * 64;

  __shared__ uint16_t Pt[16 * 32];   // softmax tile C->A re-layout
  __shared__ uint16_t Vt[64 * 32];   // V^T staging for contiguous B fragments

  Frag16 qa[2];                      // Q as two 16x32 A fragments (d=0..31, 32..63)
  {
    const uint16_t* qp = Qb + (size_t)(t0 + lrow) * 64 + lhi * 8;
    qa[0].q[0] = *(const v4u*)(qp);
    qa[0].q[1] = *(const v4u*)(qp + 16);
    qa[1].q[0] = *(const v4u*)(qp + 32);
    qa[1].q[1] = *(const v4u*)(qp + 48);
  }

  float run_m[8], run_l[8];
  v8f acc[4];
  #pragma unroll
  for (int r = 0; r < 8; ++r) { run_m[r] = -1e30f; run_l[r] = 0.f; }
  #pragma unroll
  for (int j = 0; j < 4; ++j) acc[j] = vzero8();

  const int kend = CAUSAL ? (t0 + 16) : S;
  for (int ks = 0; ks < kend; ks += 32) {
    // K as B fragments: [key half][d half]
    Frag16 kf[2][2];
    #pragma unroll
    for (int nh = 0; nh < 2; ++nh) {
      int krow = ks + nh * 16 + lrow;
      if (krow >= S) krow = S - 1;           // masked anyway
      const uint16_t* kp = Kb + (size_t)krow * 64 + lhi * 8;
      kf[nh][0].q[0] = *(const v4u*)(kp);
      kf[nh][0].q[1] = *(const v4u*)(kp + 16);
      kf[nh][1].q[0] = *(const v4u*)(kp + 32);
      kf[nh][1].q[1] = *(const v4u*)(kp + 48);
    }
    v8f s0 = vzero8(), s1 = vzero8();
    s0 = wmma_bf16(qa[0].v, kf[0][0].v, s0);
    s0 = wmma_bf16(qa[1].v, kf[0][1].v, s0);
    s1 = wmma_bf16(qa[0].v, kf[1][0].v, s1);
    s1 = wmma_bf16(qa[1].v, kf[1][1].v, s1);

    const int col0 = ks + lrow;
    const int col1 = ks + 16 + lrow;
    bool ok0 = true, ok1 = true;
    if (!CAUSAL) {
      ok0 = (cmask[(size_t)b * S + col0] != 0);
      ok1 = (cmask[(size_t)b * S + col1] != 0);
    }
    #pragma unroll
    for (int r = 0; r < 8; ++r) {
      const int mrow = t0 + lhi * 8 + r;
      float v0 = s0[r] * rscale;
      float v1 = s1[r] * rscale;
      if (CAUSAL) { if (col0 > mrow) v0 = -1e30f; if (col1 > mrow) v1 = -1e30f; }
      else        { if (!ok0)        v0 = -1e30f; if (!ok1)        v1 = -1e30f; }
      float mx = fmaxf(v0, v1);
      #pragma unroll
      for (int off = 1; off < 16; off <<= 1) mx = fmaxf(mx, __shfl_xor(mx, off, 32));
      const float nm = fmaxf(run_m[r], mx);
      const float cf = __expf(run_m[r] - nm);
      const float e0 = __expf(v0 - nm);
      const float e1 = __expf(v1 - nm);
      float rs = e0 + e1;
      #pragma unroll
      for (int off = 1; off < 16; off <<= 1) rs += __shfl_xor(rs, off, 32);
      run_l[r] = run_l[r] * cf + rs;
      run_m[r] = nm;
      #pragma unroll
      for (int j = 0; j < 4; ++j) acc[j][r] = acc[j][r] * cf;
      const int prow = lhi * 8 + r;
      Pt[prow * 32 + lrow]      = f32_to_bf16(e0);
      Pt[prow * 32 + 16 + lrow] = f32_to_bf16(e1);
    }
    // stage V^T: Vt[hd][key_local]
    {
      int key = ks + lane;
      if (key >= S) key = S - 1;
      const uint16_t* vp = Vb + (size_t)key * 64;
      #pragma unroll
      for (int c8 = 0; c8 < 64; c8 += 8) {
        v4u d = *(const v4u*)(vp + c8);
        uint16_t tmp[8];
        *(v4u*)tmp = d;
        #pragma unroll
        for (int e = 0; e < 8; ++e) Vt[(c8 + e) * 32 + lane] = tmp[e];
      }
    }
    asm volatile("s_wait_dscnt 0" ::: "memory");   // LDS stores visible to reads
    Frag16 pf;                                     // P as 16x32 A fragment
    {
      const uint16_t* pp = &Pt[lrow * 32 + lhi * 8];
      pf.q[0] = *(const v4u*)(pp);
      pf.q[1] = *(const v4u*)(pp + 16);
    }
    #pragma unroll
    for (int j = 0; j < 4; ++j) {                  // O += P * V  (hd 16-columns)
      Frag16 vf;
      const uint16_t* vpp = &Vt[(j * 16 + lrow) * 32 + lhi * 8];
      vf.q[0] = *(const v4u*)(vpp);
      vf.q[1] = *(const v4u*)(vpp + 16);
      acc[j] = wmma_bf16(pf.v, vf.v, acc[j]);
    }
  }
  #pragma unroll
  for (int r = 0; r < 8; ++r) {
    const float inv = (run_l[r] > 0.f) ? (1.f / run_l[r]) : 0.f;
    const int t = t0 + lhi * 8 + r;
    #pragma unroll
    for (int j = 0; j < 4; ++j) {
      const int col = h * 64 + j * 16 + lrow;
      Out[((size_t)b * T + t) * 1024 + col] = f32_to_bf16(acc[j][r] * inv);
    }
  }
}

// ---------- host ----------
extern "C" void kernel_launch(void* const* d_in, const int* in_sizes, int n_in,
                              void* d_out, int out_size, void* d_ws, size_t ws_size,
                              hipStream_t stream) {
  (void)in_sizes; (void)n_in; (void)out_size; (void)ws_size;
  const float* x     = (const float*)d_in[0];
  const float* cosp  = (const float*)d_in[1];
  const float* sinp  = (const float*)d_in[2];
  const float* ctx   = (const float*)d_in[3];
  const int*   cmask = (const int*)d_in[4];
  const float* w_qkv = (const float*)d_in[5];
  const float* w_out = (const float*)d_in[6];
  const float* w_q   = (const float*)d_in[7];
  const float* w_kv  = (const float*)d_in[8];
  const float* w_cout= (const float*)d_in[9];
  const float* w1    = (const float*)d_in[10];
  const float* w3    = (const float*)d_in[11];
  const float* w2    = (const float*)d_in[12];
  const float* g1    = (const float*)d_in[13];
  const float* gcr   = (const float*)d_in[14];
  const float* g2    = (const float*)d_in[15];

  const int B = 2, T = 2048, S = 1024;
  const size_t nBT_D = (size_t)B * T * 1024;   // 4 Mi
  const size_t nBS_D = (size_t)B * S * 1024;   // 2 Mi
  const size_t nQKV  = (size_t)B * T * 3072;   // 12 Mi
  const size_t nKV   = (size_t)B * S * 2048;   // 4 Mi
  const size_t nFF   = (size_t)B * T * 4096;   // 16 Mi

  char* p = (char*)d_ws;
  auto carve = [&](size_t bytes) { char* r = p; p += (bytes + 255) & ~(size_t)255; return r; };
  uint16_t* wqkv_b  = (uint16_t*)carve(3072ull * 1024 * 2);
  uint16_t* wout_b  = (uint16_t*)carve(1024ull * 1024 * 2);
  uint16_t* wq_b    = (uint16_t*)carve(1024ull * 1024 * 2);
  uint16_t* wkv_b   = (uint16_t*)carve(2048ull * 1024 * 2);
  uint16_t* wcout_b = (uint16_t*)carve(1024ull * 1024 * 2);
  uint16_t* w1_b    = (uint16_t*)carve(4096ull * 1024 * 2);
  uint16_t* w3_b    = (uint16_t*)carve(4096ull * 1024 * 2);
  uint16_t* w2_b    = (uint16_t*)carve(1024ull * 4096 * 2);
  uint16_t* ctx_b   = (uint16_t*)carve(nBS_D * 2);
  uint16_t* h_b     = (uint16_t*)carve(nBT_D * 2);
  float*    big_f   = (float*)carve(nQKV * 4);         // qkv / (q_cross + kv)
  uint16_t* q_b     = (uint16_t*)carve(nBT_D * 2);
  uint16_t* k_b     = (uint16_t*)carve(nBT_D * 2);
  uint16_t* v_b     = (uint16_t*)carve(nBT_D * 2);
  uint16_t* attn_b  = (uint16_t*)carve(nBT_D * 2);
  float*    x1      = (float*)carve(nBT_D * 4);
  float*    x2      = (float*)carve(nBT_D * 4);
  float*    a_f     = (float*)carve(nFF * 4);
  uint16_t* ff_b    = (uint16_t*)carve(nFF * 2);
  float*    q_cf    = big_f;                 // cross-q f32 (4 Mi)
  float*    kv_f    = big_f + nBT_D;         // kv f32 (4 Mi)

  auto cvt = [&](const float* s, uint16_t* d, size_t n) {
    k_f32_to_bf16<<<dim3((unsigned)((n + 255) / 256)), dim3(256), 0, stream>>>(s, d, n);
  };
  // weight + context conversion (deterministic every call)
  cvt(w_qkv,  wqkv_b,  3072ull * 1024);
  cvt(w_out,  wout_b,  1024ull * 1024);
  cvt(w_q,    wq_b,    1024ull * 1024);
  cvt(w_kv,   wkv_b,   2048ull * 1024);
  cvt(w_cout, wcout_b, 1024ull * 1024);
  cvt(w1,     w1_b,    4096ull * 1024);
  cvt(w3,     w3_b,    4096ull * 1024);
  cvt(w2,     w2_b,    1024ull * 4096);
  cvt(ctx,    ctx_b,   nBS_D);

  const float rscale = 0.125f;   // 1/sqrt(64)
  const dim3 blk256(256), blk32(32);

  // ---- self attention ----
  k_rmsnorm_bf16<<<dim3(B * T), blk256, 0, stream>>>(x, g1, h_b);
  k_gemm_bf16<0><<<dim3(3072 / 128, (B * T) / 128), blk256, 0, stream>>>(
      h_b, wqkv_b, big_f, nullptr, nullptr, B * T, 3072, 1024);
  k_rope_split<<<dim3((unsigned)((nQKV + 255) / 256)), blk256, 0, stream>>>(
      big_f, cosp, sinp, q_b, k_b, v_b, T, nQKV);
  k_attn_wmma<true><<<dim3(B * 16 * (T / 16)), blk32, 0, stream>>>(
      q_b, k_b, v_b, nullptr, attn_b, T, T, rscale);
  k_gemm_bf16<1><<<dim3(1024 / 128, (B * T) / 128), blk256, 0, stream>>>(
      attn_b, wout_b, x1, nullptr, x, B * T, 1024, 1024);

  // ---- cross attention ----
  k_rmsnorm_bf16<<<dim3(B * T), blk256, 0, stream>>>(x1, gcr, h_b);
  k_gemm_bf16<0><<<dim3(1024 / 128, (B * T) / 128), blk256, 0, stream>>>(
      h_b, wq_b, q_cf, nullptr, nullptr, B * T, 1024, 1024);
  k_split_heads<<<dim3((unsigned)((nBT_D + 255) / 256)), blk256, 0, stream>>>(q_cf, q_b, T, nBT_D);
  k_gemm_bf16<0><<<dim3(2048 / 128, (B * S) / 128), blk256, 0, stream>>>(
      ctx_b, wkv_b, kv_f, nullptr, nullptr, B * S, 2048, 1024);
  k_split_kv<<<dim3((unsigned)((nKV + 255) / 256)), blk256, 0, stream>>>(kv_f, k_b, v_b, S, nKV);
  k_attn_wmma<false><<<dim3(B * 16 * (T / 16)), blk32, 0, stream>>>(
      q_b, k_b, v_b, cmask, attn_b, T, S, rscale);
  k_gemm_bf16<1><<<dim3(1024 / 128, (B * T) / 128), blk256, 0, stream>>>(
      attn_b, wcout_b, x2, nullptr, x1, B * T, 1024, 1024);

  // ---- SwiGLU FFN ----
  k_rmsnorm_bf16<<<dim3(B * T), blk256, 0, stream>>>(x2, g2, h_b);
  k_gemm_bf16<0><<<dim3(4096 / 128, (B * T) / 128), blk256, 0, stream>>>(
      h_b, w1_b, a_f, nullptr, nullptr, B * T, 4096, 1024);
  k_gemm_bf16<2><<<dim3(4096 / 128, (B * T) / 128), blk256, 0, stream>>>(
      h_b, w3_b, nullptr, ff_b, a_f, B * T, 4096, 1024);
  k_gemm_bf16<1><<<dim3(1024 / 128, (B * T) / 128), blk256, 0, stream>>>(
      ff_b, w2_b, (float*)d_out, nullptr, x2, B * T, 1024, 4096);
}
